// MultiHeadAttention_42545946034285
// MI455X (gfx1250) — compile-verified, matrix operand order checked by reading, not run on hardware
//
#include <hip/hip_runtime.h>

typedef __bf16 bf16_t;
typedef __attribute__((ext_vector_type(16))) __bf16 v16bf;
typedef __attribute__((ext_vector_type(8)))  float  v8f;
typedef int i32x4 __attribute__((vector_size(16)));   // matches builtin's V4i

#define B_   8
#define S_   1024
#define D_   768
#define H_   12
#define DH_  64
#define MTOT (B_ * S_)   // 8192

#define AS1 __attribute__((address_space(1)))
#define AS3 __attribute__((address_space(3)))

// gfx1250 async global->LDS path (ASYNCcnt); fall back to VGPR round-trip if
// this toolchain doesn't declare the builtins.
#if __has_builtin(__builtin_amdgcn_global_load_async_to_lds_b128) && \
    __has_builtin(__builtin_amdgcn_s_wait_asynccnt)
#define ASYNC_OK 1
#else
#define ASYNC_OK 0
#endif

// Copy 16 bytes (8 bf16) global -> LDS, per lane.
__device__ __forceinline__ void stage16(bf16_t* lds, const bf16_t* g) {
#if ASYNC_OK
    __builtin_amdgcn_global_load_async_to_lds_b128(
        (AS1 i32x4*)g, (AS3 i32x4*)lds, /*offset=*/0, /*cpol=*/0);
#else
    *reinterpret_cast<uint4*>(lds) = *reinterpret_cast<const uint4*>(g);
#endif
}

__device__ __forceinline__ void wait_stage() {
#if ASYNC_OK
    __builtin_amdgcn_s_wait_asynccnt(0);
#endif
}

// ---------------------------------------------------------------------------
// Fragment loader: 16 rows x 32 K of bf16, rows K-contiguous in LDS.
// Per-lane layout for V_WMMA_*_16X16X32_BF16 (A and B operands):
//   lane<16 : row = lane,    K = {0..7, 16..23}
//   lane>=16: row = lane-16, K = {8..15, 24..31}
// ---------------------------------------------------------------------------
template<int LDW>
__device__ __forceinline__ v16bf load_frag16(const bf16_t* base) {
    const int lane = threadIdx.x & 31;
    const int idx  = lane & 15;
    const int hf   = lane >> 4;
    const bf16_t* row = base + idx * LDW + hf * 8;
    union { v16bf v; uint4 u[2]; } f;
    f.u[0] = *reinterpret_cast<const uint4*>(row);
    f.u[1] = *reinterpret_cast<const uint4*>(row + 16);
    return f.v;
}

__device__ __forceinline__ v8f wmma_bf16(v16bf a, v16bf b, v8f c) {
    return __builtin_amdgcn_wmma_f32_16x16x32_bf16(
        /*neg_a=*/false, a, /*neg_b=*/false, b,
        /*c_mod=*/(short)0, c, /*reuse_a=*/false, /*reuse_b=*/false);
}

// ---------------------------------------------------------------------------
// Pack kernels: fp32 -> bf16, weights transposed to [N][K] (K-contiguous).
// ---------------------------------------------------------------------------
__global__ void pack_f32_bf16(const float* __restrict__ in,
                              bf16_t* __restrict__ out, int n) {
    int i = blockIdx.x * 256 + threadIdx.x;
    if (i < n) out[i] = (bf16_t)in[i];
}

// W: [H, D, DH] ;  outT[n*768 + d] = W[h, d, e] with n = h*64 + e
__global__ void pack_headW(const float* __restrict__ W,
                           bf16_t* __restrict__ outT) {
    int o = blockIdx.x * 256 + threadIdx.x;
    if (o >= D_ * D_) return;
    int n = o / D_;
    int d = o % D_;
    int h = n >> 6;
    int e = n & 63;
    outT[o] = (bf16_t)W[((size_t)h * D_ + d) * DH_ + e];
}

// Wo: [D, D] row-major ; outT[n*768 + d] = Wo[d, n]
__global__ void pack_WoT(const float* __restrict__ Wo,
                         bf16_t* __restrict__ outT) {
    int o = blockIdx.x * 256 + threadIdx.x;
    if (o >= D_ * D_) return;
    int n = o / D_;
    int d = o % D_;
    outT[o] = (bf16_t)Wo[(size_t)d * D_ + n];
}

// ---------------------------------------------------------------------------
// GEMM: C[8192 x 768] = A[8192 x 768] * BT^T + bias
//   A row-major bf16 (K contiguous), BT = B transposed [N][K] bf16.
//   Block tile 128x64, BK=64, 8 waves, each wave -> 16 rows x 64 cols.
//   MODE 0: out bf16 row-major  (QKV path)
//   MODE 1: out f32  row-major  (final projection -> d_out)
// ---------------------------------------------------------------------------
template<int MODE>
__global__ __launch_bounds__(256)
void gemm_bf16_128x64(const bf16_t* __restrict__ A,
                      const bf16_t* __restrict__ BT,
                      const float* __restrict__ bias,
                      void* __restrict__ outp) {
    constexpr int LDA = 72;                 // padded row stride (bf16 elems)
    __shared__ bf16_t As[128 * LDA];        // 128 x 64 tile
    __shared__ bf16_t Bs[64 * LDA];         //  64 x 64 tile

    const int m0   = blockIdx.x * 128;
    const int n0   = blockIdx.y * 64;
    const int wave = threadIdx.x >> 5;
    const int lane = threadIdx.x & 31;
    const int idx  = lane & 15;
    const int hf   = lane >> 4;

    v8f acc[4];
#pragma unroll
    for (int j = 0; j < 4; ++j)
#pragma unroll
        for (int r = 0; r < 8; ++r) acc[j][r] = 0.0f;

    for (int kt = 0; kt < D_ / 64; ++kt) {
        const int k0 = kt * 64;
        // stage A tile: 128x64 -> 1024 16B chunks, 4 per thread
#pragma unroll
        for (int i = 0; i < 4; ++i) {
            int v = threadIdx.x + i * 256;
            int row = v >> 3, cv = v & 7;
            stage16(&As[row * LDA + cv * 8],
                    &A[(size_t)(m0 + row) * D_ + k0 + cv * 8]);
        }
        // stage B tile: 64x64 -> 512 16B chunks, 2 per thread
#pragma unroll
        for (int i = 0; i < 2; ++i) {
            int v = threadIdx.x + i * 256;
            int row = v >> 3, cv = v & 7;
            stage16(&Bs[row * LDA + cv * 8],
                    &BT[(size_t)(n0 + row) * D_ + k0 + cv * 8]);
        }
        wait_stage();
        __syncthreads();

#pragma unroll
        for (int kc = 0; kc < 64; kc += 32) {
            v16bf a = load_frag16<LDA>(&As[wave * 16 * LDA + kc]);
#pragma unroll
            for (int j = 0; j < 4; ++j) {
                v16bf b = load_frag16<LDA>(&Bs[j * 16 * LDA + kc]);
                acc[j] = wmma_bf16(a, b, acc[j]);
            }
        }
        __syncthreads();
    }

    // epilogue: C element (m = m0 + wave*16 + hf*8 + r, n = n0 + j*16 + idx)
#pragma unroll
    for (int j = 0; j < 4; ++j) {
        const int n  = n0 + j * 16 + idx;
        const float bn = bias[n];
#pragma unroll
        for (int r = 0; r < 8; ++r) {
            const int m = m0 + wave * 16 + hf * 8 + r;
            const float val = acc[j][r] + bn;
            if constexpr (MODE == 0)
                ((bf16_t*)outp)[(size_t)m * D_ + n] = (bf16_t)val;
            else
                ((float*)outp)[(size_t)m * D_ + n] = val;
        }
    }
}

// ---------------------------------------------------------------------------
// Flash-style attention. Q/K/V bf16 row-major [B*S][768], head slice at
// column h*64. One block = one (b,h) x 128 query rows; key tiles of 64.
// Each wave owns 16 query rows (4 accumulator tiles across DH=64).
// ---------------------------------------------------------------------------
__global__ __launch_bounds__(256)
void attn_kernel(const bf16_t* __restrict__ Q, const bf16_t* __restrict__ K,
                 const bf16_t* __restrict__ V, bf16_t* __restrict__ O) {
    constexpr int LD = 72;                  // padded row stride (bf16 elems)
    __shared__ bf16_t Qs[128 * LD];         // 128 q-rows x 64
    __shared__ bf16_t Ks[64 * LD];          // 64 k-rows  x 64 (rows t, cols e)
    __shared__ bf16_t VTs[64 * LD];         // V transposed (rows e, cols t)
    __shared__ bf16_t Ps[8 * 16 * LD];      // per-wave P tile 16 x 64

    const int bh   = blockIdx.y;
    const int b    = bh / H_;
    const int h    = bh % H_;
    const int q0   = blockIdx.x * 128;
    const int wave = threadIdx.x >> 5;
    const int lane = threadIdx.x & 31;
    const int idx  = lane & 15;
    const int hf   = lane >> 4;
    const size_t base = (size_t)b * S_ * D_ + (size_t)h * DH_;

    // stage Q tile once: 128x64 -> 1024 16B chunks, 4 per thread (async)
#pragma unroll
    for (int i = 0; i < 4; ++i) {
        int v = threadIdx.x + i * 256;
        int row = v >> 3, cv = v & 7;
        stage16(&Qs[row * LD + cv * 8],
                &Q[base + (size_t)(q0 + row) * D_ + cv * 8]);
    }

    v8f acc[4];
    float mrow[8], lrow[8];
#pragma unroll
    for (int j = 0; j < 4; ++j)
#pragma unroll
        for (int r = 0; r < 8; ++r) acc[j][r] = 0.0f;
#pragma unroll
    for (int r = 0; r < 8; ++r) { mrow[r] = -3.0e38f; lrow[r] = 0.0f; }

    for (int kt = 0; kt < S_ / 64; ++kt) {
        const int t0 = kt * 64;
        // stage K tile (as-is: rows t, K-contiguous e -> B operand of Q*K^T)
#pragma unroll
        for (int i = 0; i < 2; ++i) {
            int v = threadIdx.x + i * 256;
            int row = v >> 3, cv = v & 7;
            stage16(&Ks[row * LD + cv * 8],
                    &K[base + (size_t)(t0 + row) * D_ + cv * 8]);
        }
        // stage V transposed: VTs[e][t] (rows e, K-contiguous t -> B of P*V)
#pragma unroll
        for (int i = 0; i < 2; ++i) {
            int v = threadIdx.x + i * 256;
            int row = v >> 3, cv = v & 7;
            union { uint4 u; bf16_t e[8]; } tmp;
            tmp.u = *reinterpret_cast<const uint4*>(&V[base + (size_t)(t0 + row) * D_ + cv * 8]);
#pragma unroll
            for (int q = 0; q < 8; ++q) VTs[(cv * 8 + q) * LD + row] = tmp.e[q];
        }
        wait_stage();        // Q (first iter) + K async copies complete
        __syncthreads();

        // scores: S = Q(16x64) * K^T -> 4 tiles of 16x16 (cols = key rows)
        v8f sc[4];
#pragma unroll
        for (int j = 0; j < 4; ++j)
#pragma unroll
            for (int r = 0; r < 8; ++r) sc[j][r] = 0.0f;
#pragma unroll
        for (int kc = 0; kc < 64; kc += 32) {
            v16bf a = load_frag16<LD>(&Qs[wave * 16 * LD + kc]);
#pragma unroll
            for (int j = 0; j < 4; ++j) {
                v16bf bk = load_frag16<LD>(&Ks[j * 16 * LD + kc]);
                sc[j] = wmma_bf16(a, bk, sc[j]);
            }
        }

        // online softmax; C layout: row = hf*8 + r, col = j*16 + idx
        float tmax[8];
#pragma unroll
        for (int r = 0; r < 8; ++r) tmax[r] = -3.0e38f;
#pragma unroll
        for (int j = 0; j < 4; ++j)
#pragma unroll
            for (int r = 0; r < 8; ++r) {
                sc[j][r] *= 0.125f;               // 1/sqrt(DH)
                tmax[r] = fmaxf(tmax[r], sc[j][r]);
            }
#pragma unroll
        for (int r = 0; r < 8; ++r)
            for (int m = 1; m < 16; m <<= 1)
                tmax[r] = fmaxf(tmax[r], __shfl_xor(tmax[r], m, 32));

        float alpha[8], tsum[8];
#pragma unroll
        for (int r = 0; r < 8; ++r) {
            float mn = fmaxf(mrow[r], tmax[r]);
            alpha[r] = __expf(mrow[r] - mn);
            mrow[r]  = mn;
            tsum[r]  = 0.0f;
        }
#pragma unroll
        for (int j = 0; j < 4; ++j)
#pragma unroll
            for (int r = 0; r < 8; ++r) {
                float p = __expf(sc[j][r] - mrow[r]);
                sc[j][r] = p;
                tsum[r] += p;
            }
#pragma unroll
        for (int r = 0; r < 8; ++r) {
            for (int m = 1; m < 16; m <<= 1) tsum[r] += __shfl_xor(tsum[r], m, 32);
            lrow[r] = lrow[r] * alpha[r] + tsum[r];
        }
#pragma unroll
        for (int j = 0; j < 4; ++j)
#pragma unroll
            for (int r = 0; r < 8; ++r) acc[j][r] *= alpha[r];

        // C-layout -> A-layout via per-wave LDS round trip (bf16 probs)
        bf16_t* Pw = &Ps[wave * 16 * LD];
#pragma unroll
        for (int j = 0; j < 4; ++j)
#pragma unroll
            for (int r = 0; r < 8; ++r)
                Pw[(hf * 8 + r) * LD + j * 16 + idx] = (bf16_t)sc[j][r];

        // O += P(16x64) * V(64x64); B operand from transposed V tile
#pragma unroll
        for (int kc = 0; kc < 64; kc += 32) {
            v16bf a = load_frag16<LD>(&Ps[wave * 16 * LD + kc]);
#pragma unroll
            for (int j = 0; j < 4; ++j) {
                v16bf bv = load_frag16<LD>(&VTs[j * 16 * LD + kc]);
                acc[j] = wmma_bf16(a, bv, acc[j]);
            }
        }
        __syncthreads();
    }

    float inv[8];
#pragma unroll
    for (int r = 0; r < 8; ++r) inv[r] = 1.0f / lrow[r];
#pragma unroll
    for (int j = 0; j < 4; ++j) {
        const int e = j * 16 + idx;
#pragma unroll
        for (int r = 0; r < 8; ++r) {
            const int s = q0 + wave * 16 + hf * 8 + r;
            O[base + (size_t)s * D_ + e] = (bf16_t)(acc[j][r] * inv[r]);
        }
    }
}

// ---------------------------------------------------------------------------
// Launch: pack -> QKV GEMMs -> attention -> output projection.
// Workspace (bf16): xb | qb | kb | vb | wqT | wkT | wvT | woT  (~53 MB).
// Attention output reuses xb (x is dead after the QKV GEMMs).
// ---------------------------------------------------------------------------
extern "C" void kernel_launch(void* const* d_in, const int* in_sizes, int n_in,
                              void* d_out, int out_size, void* d_ws, size_t ws_size,
                              hipStream_t stream) {
    (void)in_sizes; (void)n_in; (void)out_size;

    const float* x  = (const float*)d_in[0];
    const float* Wq = (const float*)d_in[1];
    const float* bq = (const float*)d_in[2];
    const float* Wk = (const float*)d_in[3];
    const float* bk = (const float*)d_in[4];
    const float* Wv = (const float*)d_in[5];
    const float* bv = (const float*)d_in[6];
    const float* Wo = (const float*)d_in[7];
    const float* bo = (const float*)d_in[8];

    const size_t MD = (size_t)MTOT * D_;   // 6,291,456 elems
    const size_t WN = (size_t)D_ * D_;     //   589,824 elems
    if (ws_size < (4 * MD + 4 * WN) * sizeof(bf16_t)) return;

    bf16_t* xb  = (bf16_t*)d_ws;
    bf16_t* qb  = xb  + MD;
    bf16_t* kb  = qb  + MD;
    bf16_t* vb  = kb  + MD;
    bf16_t* wqT = vb  + MD;
    bf16_t* wkT = wqT + WN;
    bf16_t* wvT = wkT + WN;
    bf16_t* woT = wvT + WN;

    pack_f32_bf16<<<(int)((MD + 255) / 256), 256, 0, stream>>>(x, xb, (int)MD);
    pack_headW<<<(int)((WN + 255) / 256), 256, 0, stream>>>(Wq, wqT);
    pack_headW<<<(int)((WN + 255) / 256), 256, 0, stream>>>(Wk, wkT);
    pack_headW<<<(int)((WN + 255) / 256), 256, 0, stream>>>(Wv, wvT);
    pack_WoT  <<<(int)((WN + 255) / 256), 256, 0, stream>>>(Wo, woT);

    dim3 gg(MTOT / 128, D_ / 64);
    gemm_bf16_128x64<0><<<gg, 256, 0, stream>>>(xb, wqT, bq, qb);
    gemm_bf16_128x64<0><<<gg, 256, 0, stream>>>(xb, wkT, bk, kb);
    gemm_bf16_128x64<0><<<gg, 256, 0, stream>>>(xb, wvT, bv, vb);

    attn_kernel<<<dim3(S_ / 128, B_ * H_), 256, 0, stream>>>(qb, kb, vb, xb);

    gemm_bf16_128x64<1><<<gg, 256, 0, stream>>>(xb, woT, bo, d_out);
}
